// ForwardEulerPEM_mimo_35768487641341
// MI455X (gfx1250) — matrix operand order repaired and checked by reading, not compiled, not run on hardware
//
#include <hip/hip_runtime.h>

// Forward-Euler rollout of x_{q+1} = x_q + (W2 tanh(W1 [x;u_q] + b1) + b2)*dt
// Strictly sequential over 32768 steps -> latency-bound; one persistent
// workgroup of 4 wave32s on one WGP (1 wave/SIMD -> max VGPR budget).
// W1: register-resident bf16 in WMMA A-layout (768 VGPRs/wave).
// W2: LDS-resident bf16 (256KB), pre-swizzled so the loop reads each A-tile
//     chunk as two ds_load_b128 per lane.

#define DT_STEP 0.01f
#define NSTATE  128
#define NINPUT  64
#define NZ      192   // NSTATE + NINPUT
#define HIDDEN  1024
#define NTHREADS 128  // 4 wave32
#define TPW     16    // hidden 16x16 M-tiles per wave
#define KSUB    8     // layer-2 K sub-chunks of 32 per wave

typedef __attribute__((ext_vector_type(16))) __bf16 v16bf;
typedef __attribute__((ext_vector_type(8)))  float  v8f;

union B16 {
    v16bf v;
    unsigned int u[8];
    unsigned short s[16];
};

__device__ __forceinline__ unsigned short f2bf(float f) {
    unsigned int x = __float_as_uint(f);
    x += 0x7FFFu + ((x >> 16) & 1u);
    return (unsigned short)(x >> 16);
}
__device__ __forceinline__ unsigned int pack2bf(float a, float b) {
    return (unsigned int)f2bf(a) | ((unsigned int)f2bf(b) << 16);
}

// Load one 16x32 A-tile chunk (this lane's 16 bf16 elements) from a row-major
// f32 matrix. A-layout K indices are {kb..kb+7, kb+16..kb+23} contiguous in
// groups of 8, so four float4 loads suffice. kb (= 32c + koff) keeps 16B align.
__device__ __forceinline__ void load_a_chunk(B16& dst, const float* rowbase,
                                             int kb) {
    const float4 w0 = *(const float4*)(rowbase + kb);
    const float4 w1 = *(const float4*)(rowbase + kb + 4);
    const float4 w2 = *(const float4*)(rowbase + kb + 16);
    const float4 w3 = *(const float4*)(rowbase + kb + 20);
    dst.u[0] = pack2bf(w0.x, w0.y);
    dst.u[1] = pack2bf(w0.z, w0.w);
    dst.u[2] = pack2bf(w1.x, w1.y);
    dst.u[3] = pack2bf(w1.z, w1.w);
    dst.u[4] = pack2bf(w2.x, w2.y);
    dst.u[5] = pack2bf(w2.z, w2.w);
    dst.u[6] = pack2bf(w3.x, w3.y);
    dst.u[7] = pack2bf(w3.z, w3.w);
}

// dynamic LDS layout (bytes)
#define OFF_W2    0          // uint w2lds[8*32*32*8]            : 262144
#define OFF_HPRE  262144     // float hpre[HIDDEN]               :   4096
#define OFF_B1    266240     // float b1buf[HIDDEN]              :   4096
#define OFF_DXP   270336     // float dxpart[4][NSTATE]          :   2048
#define OFF_X     272384     // float xbuf[NSTATE]               :    512
#define OFF_B2    272896     // float b2buf[NSTATE]              :    512
#define OFF_H     273408     // ushort hbuf[HIDDEN]              :   2048
#define OFF_Z     275456     // ushort zbuf[NZ] (padded)         :    512
#define SMEM_BYTES 275968

__global__ __launch_bounds__(NTHREADS, 1)
void ForwardEulerPEM_rollout_kernel(const float* __restrict__ x0,
                                    const float* __restrict__ u,
                                    const float* __restrict__ W1,
                                    const float* __restrict__ b1,
                                    const float* __restrict__ W2,
                                    const float* __restrict__ b2,
                                    float* __restrict__ out,
                                    int n_steps)
{
    extern __shared__ __align__(16) unsigned char smem[];
    unsigned int*   w2lds  = (unsigned int*)(smem + OFF_W2);
    float*          hpre   = (float*)(smem + OFF_HPRE);
    float*          b1buf  = (float*)(smem + OFF_B1);
    float*          dxpart = (float*)(smem + OFF_DXP);
    float*          xbuf   = (float*)(smem + OFF_X);
    float*          b2buf  = (float*)(smem + OFF_B2);
    unsigned short* hbuf   = (unsigned short*)(smem + OFF_H);
    unsigned short* zbuf   = (unsigned short*)(smem + OFF_Z);

    const int tid  = threadIdx.x;          // 0..127
    const int wave = tid >> 5;             // 0..3
    const int lane = tid & 31;
    const bool hi  = lane >= 16;
    const int lrow = lane & 15;            // A/C row within a 16-tile
    const int koff = hi ? 8 : 0;           // 16-bit A-layout K sub-offset

    // ---------------- prologue: small LDS state ----------------
    for (int i = tid; i < HIDDEN; i += NTHREADS) b1buf[i] = b1[i];
    {
        float x = x0[tid];                 // tid < 128 == NSTATE always
        xbuf[tid]  = x;
        b2buf[tid] = b2[tid];
        zbuf[tid]  = f2bf(x);
    }
    if (tid < NINPUT) zbuf[NSTATE + tid] = f2bf(u[tid]);   // u[0]

    // ---------------- prologue: W1 -> VGPRs (bf16, WMMA A-layout) -------
    B16 a1[TPW][6];
#pragma unroll
    for (int t = 0; t < TPW; ++t) {
        const int row = (wave * TPW + t) * 16 + lrow;
        const float* wr = W1 + row * NZ;
#pragma unroll
        for (int c = 0; c < 6; ++c)
            load_a_chunk(a1[t][c], wr, 32 * c + koff);
    }

    // ---------------- prologue: W2 -> LDS, pre-swizzled A-layout --------
    // w2lds[(((m*32 + kc)*32) + lane)*8 + v]: this lane's 8 uints contiguous
    // (32B) -> the loop reads each chunk as two ds_load_b128.
    for (int m = 0; m < 8; ++m) {
        for (int j = 0; j < KSUB; ++j) {
            const int kc = wave * KSUB + j;
            B16 a;
            load_a_chunk(a, W2 + (m * 16 + lrow) * HIDDEN,
                         256 * wave + 32 * j + koff);
            unsigned int* dstp = &w2lds[((m * 32 + kc) * 32 + lane) * 8];
            *(uint4*)(dstp)     = make_uint4(a.u[0], a.u[1], a.u[2], a.u[3]);
            *(uint4*)(dstp + 4) = make_uint4(a.u[4], a.u[5], a.u[6], a.u[7]);
        }
    }

    __syncthreads();

    // ---------------- sequential rollout ----------------
    for (int q = 0; q < n_steps; ++q) {
        // stage-ahead: u[q+1] into registers (latency hidden behind layer 1)
        float unext = 0.0f;
        if (tid < NINPUT && q + 1 < n_steps)
            unext = u[(size_t)(q + 1) * NINPUT + tid];
        if (tid == NINPUT && q + 32 < n_steps)
            __builtin_prefetch(u + (size_t)(q + 32) * NINPUT, 0, 1);

        // B operand for layer 1: z broadcast into all 16 columns.
        // 16-bit B layout (matches A): lanes 0-15 K=0-7,16-23, lanes 16-31 K=8-15,24-31, 2 K/VGPR.
        B16 bz[6];
#pragma unroll
        for (int c = 0; c < 6; ++c) {
#pragma unroll
            for (int v = 0; v < 8; ++v) {
                const int k = 32 * c + koff + 2 * v + (v >= 4 ? 8 : 0); // even -> 4B ok
                bz[c].u[v] = *(const unsigned int*)&zbuf[k];
            }
        }

        // ---- layer 1: 16 tiles x 6 K-chunks, W1 from registers
#pragma unroll
        for (int t = 0; t < TPW; ++t) {
            v8f acc = (v8f){};
#pragma unroll
            for (int c = 0; c < 6; ++c) {
                acc = __builtin_amdgcn_wmma_f32_16x16x32_bf16(
                    false, a1[t][c].v, false, bz[c].v, (short)0, acc,
                    false, false);
            }
            // columns identical; lanes 0/16 carry M=r / M=r+8
            if (lane == 0 || lane == 16) {
                const int base = (wave * TPW + t) * 16 + koff;
                *(float4*)&hpre[base] =
                    make_float4(acc[0], acc[1], acc[2], acc[3]);
                *(float4*)&hpre[base + 4] =
                    make_float4(acc[4], acc[5], acc[6], acc[7]);
            }
        }
        __syncthreads();

        // ---- bias + tanh distributed over all 128 threads (8 rows each)
        {
            const int i0 = tid * 8;
            const float4 p0 = *(const float4*)&hpre[i0];
            const float4 p1 = *(const float4*)&hpre[i0 + 4];
            const float4 g0 = *(const float4*)&b1buf[i0];
            const float4 g1 = *(const float4*)&b1buf[i0 + 4];
            uint4 pk;
            pk.x = pack2bf(tanhf(p0.x + g0.x), tanhf(p0.y + g0.y));
            pk.y = pack2bf(tanhf(p0.z + g0.z), tanhf(p0.w + g0.w));
            pk.z = pack2bf(tanhf(p1.x + g1.x), tanhf(p1.y + g1.y));
            pk.w = pack2bf(tanhf(p1.z + g1.z), tanhf(p1.w + g1.w));
            *(uint4*)&hbuf[i0] = pk;
            // zbuf's B-operand reads are done; stage next-step u part now
            if (tid < NINPUT && q + 1 < n_steps)
                zbuf[NSTATE + tid] = f2bf(unext);
        }
        __syncthreads();

        // ---- layer 2: dx_part = W2[:, 256w:256w+256) @ h[256w:256w+256)
        B16 bh[KSUB];
#pragma unroll
        for (int j = 0; j < KSUB; ++j) {
#pragma unroll
            for (int v = 0; v < 8; ++v) {
                const int k = 256 * wave + 32 * j + koff + 2 * v + (v >= 4 ? 8 : 0);
                bh[j].u[v] = *(const unsigned int*)&hbuf[k];
            }
        }
#pragma unroll
        for (int m = 0; m < 8; ++m) {
            v8f d = (v8f){};
#pragma unroll
            for (int j = 0; j < KSUB; ++j) {
                const int kc = wave * KSUB + j;
                const unsigned int* srcp =
                    &w2lds[((m * 32 + kc) * 32 + lane) * 8];
                const uint4 A0 = *(const uint4*)(srcp);
                const uint4 A1 = *(const uint4*)(srcp + 4);
                B16 a;
                a.u[0] = A0.x; a.u[1] = A0.y; a.u[2] = A0.z; a.u[3] = A0.w;
                a.u[4] = A1.x; a.u[5] = A1.y; a.u[6] = A1.z; a.u[7] = A1.w;
                d = __builtin_amdgcn_wmma_f32_16x16x32_bf16(
                    false, a.v, false, bh[j].v, (short)0, d, false, false);
            }
            if (lane == 0 || lane == 16) {
                const int base = wave * NSTATE + m * 16 + koff;
                *(float4*)&dxpart[base] = make_float4(d[0], d[1], d[2], d[3]);
                *(float4*)&dxpart[base + 4] =
                    make_float4(d[4], d[5], d[6], d[7]);
            }
        }
        __syncthreads();

        // ---- deterministic 4-way reduction + Euler update + output
        {
            const float dx = dxpart[0 * NSTATE + tid] +
                             dxpart[1 * NSTATE + tid] +
                             dxpart[2 * NSTATE + tid] +
                             dxpart[3 * NSTATE + tid];
            const float xn = xbuf[tid] + (dx + b2buf[tid]) * DT_STEP;
            xbuf[tid] = xn;
            zbuf[tid] = f2bf(xn);
            __builtin_nontemporal_store(xn, &out[(size_t)q * NSTATE + tid]);
        }
        __syncthreads();
    }
}

extern "C" void kernel_launch(void* const* d_in, const int* in_sizes, int n_in,
                              void* d_out, int out_size, void* d_ws, size_t ws_size,
                              hipStream_t stream) {
    (void)n_in; (void)out_size; (void)d_ws; (void)ws_size;
    const float* x0 = (const float*)d_in[0];
    const float* u  = (const float*)d_in[1];
    // d_in[2] = y, unused in update==0 mode
    const float* W1 = (const float*)d_in[3];
    const float* b1 = (const float*)d_in[4];
    const float* W2 = (const float*)d_in[5];
    const float* b2 = (const float*)d_in[6];
    float* out = (float*)d_out;

    const int n_steps = in_sizes[1] / NINPUT;

    ForwardEulerPEM_rollout_kernel<<<1, NTHREADS, SMEM_BYTES, stream>>>(
        x0, u, W1, b1, W2, b2, out, n_steps);
}
